// FD_encoder_38268158607797
// MI455X (gfx1250) — matrix operand order repaired and checked by reading, not compiled, not run on hardware
//
#include <hip/hip_runtime.h>
#include <math.h>

// ---------------------------------------------------------------------------
// FD_encoder for MI455X (gfx1250, wave32, WMMA).
// Pipeline: 5x TCN -> scatter into d[N,96] -> (3x) radix-select top-half edges
// -> 3-layer GatedGraphConv (WMMA f16 GEMMs + atomic mean-aggregation)
// -> attention over {int,int,res,lat} heads -> Wo -> (z_mu, softplus z_var).
// h_invoke / invoke edges are dead code in the reference and are skipped.
// GEMM: block = 8 waves = 128x64 C block; B panel staged in LDS via
// global_load_async_to_lds_b128 (ASYNCcnt) when available.
// ---------------------------------------------------------------------------

#define NUMN 170
#define BATCH 64
#define NN (NUMN*BATCH)      // 10880 nodes
#define GN (NN/5)            // 2176 per resource type
#define LT 30
#define CO 32
#define KSZ 7
#define EMB 64
#define GG 128
#define ZD 64
#define EE (16*NN)           // 174080 edges
#define KEEPN (EE/2)         // 87040 kept

typedef __attribute__((ext_vector_type(16))) _Float16 v16h;
typedef __attribute__((ext_vector_type(8)))  float    v8f;
typedef __attribute__((ext_vector_type(2)))  float    v2f;

// async-to-LDS builtin takes int4-vector pointers (AS1 src, AS3 dst)
typedef int v4i __attribute__((vector_size(16)));
typedef __attribute__((address_space(1))) v4i gas_v4i;
typedef __attribute__((address_space(3))) v4i las_v4i;

#if defined(__HIP_DEVICE_COMPILE__) && \
    __has_builtin(__builtin_amdgcn_global_load_async_to_lds_b128) && \
    __has_builtin(__builtin_amdgcn_s_wait_asynccnt)
#define USE_ASYNC_LDS 1
#else
#define USE_ASYNC_LDS 0
#endif

// ---------------- utility kernels ----------------

__global__ void k_fill_f32(float* p, int n, float v){
  int i = blockIdx.x*256 + threadIdx.x;
  if (i < n) p[i] = v;
}

__global__ void k_transpose(const float* __restrict__ src, float* __restrict__ dst,
                            int rows, int cols){
  int i = blockIdx.x*256 + threadIdx.x;
  if (i >= rows*cols) return;
  int r = i / cols, c = i % cols;
  dst[c*rows + r] = src[i];
}

// ---------------- TCN (2-layer dilated causal conv, last timestep) ---------
// Layer2 (dil=2,K=7) at t=29 needs h1 at t = 17,19,...,29 only.

__global__ void k_tcn(const float* __restrict__ x, int cin,
                      const float* __restrict__ w1, const float* __restrict__ b1,
                      const float* __restrict__ w2, const float* __restrict__ b2,
                      const int* __restrict__ ni, const float* __restrict__ embed,
                      float* __restrict__ dfeat){
  __shared__ float h1[KSZ][CO];
  int g  = blockIdx.x;
  int co = threadIdx.x;            // 0..31, one wave per node
  for (int k = 0; k < KSZ; ++k){
    int t = 17 + 2*k;
    float acc = b1[co];
    for (int ci = 0; ci < cin; ++ci){
      const float* xr = x + (size_t)(g*cin + ci)*LT + (t - 6);
      const float* wr = w1 + (size_t)(co*cin + ci)*KSZ;
      #pragma unroll
      for (int kk = 0; kk < KSZ; ++kk) acc += wr[kk]*xr[kk];
    }
    h1[k][co] = fmaxf(acc, 0.f);
  }
  __syncthreads();
  float acc = b2[co];
  for (int c = 0; c < CO; ++c){
    const float* wr = w2 + (size_t)(co*CO + c)*KSZ;
    #pragma unroll
    for (int kk = 0; kk < KSZ; ++kk) acc += wr[kk]*h1[kk][c];
  }
  int node = ni[g];
  float* dr = dfeat + (size_t)node*96;
  dr[co] = fmaxf(acc, 0.f);
  int eb = node % NUMN;
  dr[32 + co] = embed[(size_t)eb*EMB + co];
  dr[64 + co] = embed[(size_t)eb*EMB + 32 + co];
}

// ---------------- structure learning: score + radix select -----------------

__global__ void k_score(const float* __restrict__ dfeat, const int* __restrict__ ei,
                        unsigned* __restrict__ keys){
  int e = blockIdx.x*256 + threadIdx.x;
  if (e >= EE) return;
  int s = ei[e], t = ei[EE + e];
  const float4* a = (const float4*)(dfeat + (size_t)s*96);
  const float4* b = (const float4*)(dfeat + (size_t)t*96);
  float acc = 0.f;
  #pragma unroll
  for (int i = 0; i < 24; ++i){
    float4 av = a[i], bv = b[i];
    acc += av.x*bv.x + av.y*bv.y + av.z*bv.z + av.w*bv.w;
  }
  unsigned u = __float_as_uint(acc);
  u = (u & 0x80000000u) ? ~u : (u | 0x80000000u);  // order-preserving key
  keys[e] = u;
}

// state: [0]=prefix [2]=kth [3]=threshold [4]=count_gt [5]=ties_needed
//        [6]=tie_taken [7]=kept_count [8..263]=hist
__global__ void k_state_init(unsigned* st){
  int i = threadIdx.x;                       // blockDim = 512
  st[i] = (i == 2) ? (unsigned)KEEPN : 0u;
}

__global__ void k_hist(const unsigned* __restrict__ keys, unsigned* st, int pass){
  __shared__ unsigned lh[256];
  int t = threadIdx.x;
  lh[t] = 0u;
  __syncthreads();
  int e = blockIdx.x*256 + t;
  if (e < EE){
    unsigned key = keys[e];
    unsigned maskHigh = (pass == 0) ? 0u : (0xFFFFFFFFu << (32 - 8*pass));
    if (((key ^ st[0]) & maskHigh) == 0u){
      int shift = 24 - 8*pass;
      atomicAdd(&lh[(key >> shift) & 255u], 1u);    // LDS atomic
    }
  }
  __syncthreads();
  unsigned c = lh[t];
  if (c) atomicAdd(&st[8 + t], c);                  // one global add per bin
}

__global__ void k_pick(unsigned* st, int pass){
  unsigned kth = st[2], cum = 0; int chosen = 0;
  for (int v = 255; v >= 0; --v){
    unsigned c = st[8+v];
    if (cum + c >= kth){ chosen = v; break; }
    cum += c;
  }
  int shift = 24 - 8*pass;
  st[0] |= ((unsigned)chosen) << shift;
  st[2] = kth - cum;
  for (int i = 0; i < 256; ++i) st[8+i] = 0u;
  if (pass == 3) st[3] = st[0];              // exact KEEP-th largest key
}

__global__ void k_count_gt(const unsigned* __restrict__ keys, unsigned* st){
  int e = blockIdx.x*256 + threadIdx.x;
  if (e >= EE) return;
  if (keys[e] > st[3]) atomicAdd(&st[4], 1u);
}

__global__ void k_finalize(unsigned* st){
  st[5] = (unsigned)KEEPN - st[4];
  st[6] = 0u; st[7] = 0u;
}

__global__ void k_select(const unsigned* __restrict__ keys, const int* __restrict__ ei,
                         unsigned* st, int* __restrict__ ksrc, int* __restrict__ kdst,
                         float* __restrict__ cnt){
  int e = blockIdx.x*256 + threadIdx.x;
  if (e >= EE) return;
  unsigned key = keys[e];
  unsigned T = st[3];
  bool keep = key > T;
  if (!keep && key == T) keep = (atomicAdd(&st[6], 1u) < st[5]);
  if (keep){
    unsigned i = atomicAdd(&st[7], 1u);
    if (i < (unsigned)KEEPN){
      int dst = ei[EE + e];
      ksrc[i] = ei[e];
      kdst[i] = dst;
      atomicAdd(&cnt[dst], 1.0f);
    }
  }
}

// ---------------- GGNN helpers ----------------

__global__ void k_init_h(const float* __restrict__ dfeat, float* __restrict__ h){
  int i = blockIdx.x*256 + threadIdx.x;
  if (i >= NN*GG) return;
  int n = i >> 7, c = i & 127;
  h[i] = (c < 96) ? dfeat[(size_t)n*96 + c] : 0.f;
}

__global__ void k_aggregate(const float* __restrict__ m, const int* __restrict__ ksrc,
                            const int* __restrict__ kdst, float* __restrict__ agg){
  int tid = blockIdx.x*256 + threadIdx.x;
  if (tid >= KEEPN*32) return;
  int e = tid >> 5, q = tid & 31;
  int s = ksrc[e], d = kdst[e];
  float4 v = *(const float4*)(m + (size_t)s*GG + q*4);
  float* ap = agg + (size_t)d*GG + q*4;
  atomicAdd(ap+0, v.x); atomicAdd(ap+1, v.y);
  atomicAdd(ap+2, v.z); atomicAdd(ap+3, v.w);
}

__global__ void k_divide(float* __restrict__ agg, const float* __restrict__ cnt){
  int i = blockIdx.x*256 + threadIdx.x;
  if (i >= NN*GG) return;
  agg[i] /= fmaxf(cnt[i >> 7], 1.0f);
}

__global__ void k_gate(const float* __restrict__ gi, const float* __restrict__ gh,
                       float* __restrict__ h){
  int i = blockIdx.x*256 + threadIdx.x;
  if (i >= NN*GG) return;
  int n = i >> 7, c = i & 127;
  const float* gir = gi + (size_t)n*384;
  const float* ghr = gh + (size_t)n*384;
  float r  = 1.f/(1.f + expf(-(gir[c]       + ghr[c])));
  float z  = 1.f/(1.f + expf(-(gir[128 + c] + ghr[128 + c])));
  float nn = tanhf(gir[256 + c] + r*ghr[256 + c]);
  h[i] = (1.f - z)*nn + z*h[i];
}

// ---------------- WMMA GEMM: C[NN,Ncols] = A[NN,128] * Bt[Ncols,128]^T -----
// Block = 256 threads = 8 waves -> 128x64 C block. Wave w owns a 16-row strip
// and 4 accumulators (16x64), reusing its A fragment across 4 B fragments per
// K-step: 16x V_WMMA_F32_16X16X32_F16 per wave. The 64x128 f32 B panel is
// staged in LDS (row stride 132 floats: conflict-free column reads) via
// global_load_async_to_lds_b128 + s_wait_asynccnt when the builtins exist.
// Fragments per ISA 7.12.2:
//  A: lane<16 row r holds K{c..c+7, c+16..c+23}, c = k0+8*(lane>=16)
//  B: lane col r holds K{c..c+7, c+16..c+23}, c = k0+8*(lane>=16)
//  C: VGPR rr -> M = rr + 8*half, N = lane&15.

#define BN 64
#define BSTR 132

__global__ __launch_bounds__(256) void k_gemm_wmma(
    const float* __restrict__ A, const float* __restrict__ Bt,
    const float* __restrict__ bias, float* __restrict__ C,
    int Ncols, int act){
  __shared__ float ldsB[BN*BSTR];
  int nblocks = Ncols >> 6;
  int mb = blockIdx.x / nblocks;
  int nb = blockIdx.x % nblocks;
  const float* bsrc = Bt + (size_t)nb*BN*128;
  int t = threadIdx.x;

  // stage B panel: 64 rows x 128 floats (2048 x 16B)
#if USE_ASYNC_LDS
  #pragma unroll
  for (int j = 0; j < 8; ++j){
    int idx = j*256 + t;
    int row = idx >> 5, chunk = idx & 31;
    __builtin_amdgcn_global_load_async_to_lds_b128(
        (gas_v4i*)(bsrc + (size_t)row*128 + chunk*4),
        (las_v4i*)(&ldsB[row*BSTR + chunk*4]),
        0, 0);
  }
  __builtin_amdgcn_s_wait_asynccnt(0);
#else
  #pragma unroll
  for (int j = 0; j < 8; ++j){
    int idx = j*256 + t;
    int row = idx >> 5, chunk = idx & 31;
    *(float4*)(&ldsB[row*BSTR + chunk*4]) =
        *(const float4*)(bsrc + (size_t)row*128 + chunk*4);
  }
#endif
  __syncthreads();

  int w = t >> 5;
  int lane = t & 31;
  int half = lane >> 4, r = lane & 15;
  const float* arow = A + (size_t)((mb*8 + w)*16 + r)*128;

  v8f acc[4] = {};
  #pragma unroll
  for (int ki = 0; ki < 4; ++ki){
    int k0 = ki*32;
    int ca = k0 + 8*half;
    float4 a0 = *(const float4*)(arow + ca);
    float4 a1 = *(const float4*)(arow + ca + 4);
    float4 a2 = *(const float4*)(arow + ca + 16);
    float4 a3 = *(const float4*)(arow + ca + 20);
    v16h va;
    va[0]=(_Float16)a0.x;  va[1]=(_Float16)a0.y;  va[2]=(_Float16)a0.z;  va[3]=(_Float16)a0.w;
    va[4]=(_Float16)a1.x;  va[5]=(_Float16)a1.y;  va[6]=(_Float16)a1.z;  va[7]=(_Float16)a1.w;
    va[8]=(_Float16)a2.x;  va[9]=(_Float16)a2.y;  va[10]=(_Float16)a2.z; va[11]=(_Float16)a2.w;
    va[12]=(_Float16)a3.x; va[13]=(_Float16)a3.y; va[14]=(_Float16)a3.z; va[15]=(_Float16)a3.w;
    int cb = k0 + 8*half;
    #pragma unroll
    for (int nt = 0; nt < 4; ++nt){
      const float* bcol = &ldsB[(nt*16 + r)*BSTR + cb];
      float4 b0 = *(const float4*)(bcol);
      float4 b1 = *(const float4*)(bcol + 4);
      float4 b2 = *(const float4*)(bcol + 16);
      float4 b3 = *(const float4*)(bcol + 20);
      v16h vb;
      vb[0]=(_Float16)b0.x;  vb[1]=(_Float16)b0.y;  vb[2]=(_Float16)b0.z;  vb[3]=(_Float16)b0.w;
      vb[4]=(_Float16)b1.x;  vb[5]=(_Float16)b1.y;  vb[6]=(_Float16)b1.z;  vb[7]=(_Float16)b1.w;
      vb[8]=(_Float16)b2.x;  vb[9]=(_Float16)b2.y;  vb[10]=(_Float16)b2.z; vb[11]=(_Float16)b2.w;
      vb[12]=(_Float16)b3.x; vb[13]=(_Float16)b3.y; vb[14]=(_Float16)b3.z; vb[15]=(_Float16)b3.w;
      acc[nt] = __builtin_amdgcn_wmma_f32_16x16x32_f16(false, va, false, vb,
                                                       (short)0, acc[nt], false, false);
    }
  }

  #pragma unroll
  for (int nt = 0; nt < 4; ++nt){
    int col = nb*BN + nt*16 + r;
    float bv = bias ? bias[col] : 0.f;
    #pragma unroll
    for (int rr = 0; rr < 8; ++rr){
      int row = (mb*8 + w)*16 + rr + 8*half;
      float v = acc[nt][rr] + bv;
      if (act == 1) v = tanhf(v);                               // tanh
      else if (act == 2) v = (v > 20.f) ? v : log1pf(expf(v));  // softplus
      C[(size_t)row*Ncols + col] = v;
    }
  }
}

// Probe kernel (confirmed to codegen last round): dense f32 WMMA 16x16x4.
#if defined(__HIP_DEVICE_COMPILE__)
#if __has_builtin(__builtin_amdgcn_wmma_f32_16x16x4_f32)
__global__ void k_probe_wmma_f32(const float* A, const float* B, float* C){
  v2f a = *(const v2f*)(A + threadIdx.x*2);
  v2f b = *(const v2f*)(B + threadIdx.x*2);
  v8f c = {};
  c = __builtin_amdgcn_wmma_f32_16x16x4_f32(false, a, false, b, (short)0, c, false, false);
  *(v8f*)(C + threadIdx.x*8) = c;
}
#endif
#endif

// ---------------- attention (heads: int,int,res,lat) -----------------------

__global__ void k_att(const float* __restrict__ t0, const float* __restrict__ t2,
                      const float* __restrict__ t3, const float* __restrict__ h0,
                      const float* __restrict__ h2, const float* __restrict__ h3,
                      const float* __restrict__ v, float* __restrict__ ctx){
  int wave = blockIdx.x*4 + (threadIdx.x >> 5);
  if (wave >= NN) return;
  int lane = threadIdx.x & 31;
  const float* r0 = t0 + (size_t)wave*GG;
  const float* r2 = t2 + (size_t)wave*GG;
  const float* r3 = t3 + (size_t)wave*GG;
  float d0 = 0.f, d2 = 0.f, d3 = 0.f;
  for (int j = lane; j < GG; j += 32){
    float vv = v[j];
    d0 += vv*r0[j]; d2 += vv*r2[j]; d3 += vv*r3[j];
  }
  for (int o = 16; o > 0; o >>= 1){
    d0 += __shfl_xor(d0, o, 32);
    d2 += __shfl_xor(d2, o, 32);
    d3 += __shfl_xor(d3, o, 32);
  }
  float mx = fmaxf(d0, fmaxf(d2, d3));
  float a0 = expf(d0 - mx), a2 = expf(d2 - mx), a3 = expf(d3 - mx);
  float inv = 1.f/(2.f*a0 + a2 + a3);       // head0==head1
  float w0 = 2.f*a0*inv, w2 = a2*inv, w3 = a3*inv;
  const float* q0 = h0 + (size_t)wave*GG;
  const float* q2 = h2 + (size_t)wave*GG;
  const float* q3 = h3 + (size_t)wave*GG;
  float* co = ctx + (size_t)wave*GG;
  for (int j = lane; j < GG; j += 32)
    co[j] = w0*q0[j] + w2*q2[j] + w3*q3[j];
}

// ---------------- host orchestration ----------------

extern "C" void kernel_launch(void* const* d_in, const int* in_sizes, int n_in,
                              void* d_out, int out_size, void* d_ws, size_t ws_size,
                              hipStream_t stream){
  (void)in_sizes; (void)n_in; (void)out_size; (void)ws_size;
  // input indices (insertion-order flatten):
  // 0..4 x, 5 invoke_attr(unused), 6..10 ni, 11 invoke_ei(unused), 12..14 ei,
  // 15 embed, 16+4i tcn{w1,b1,w2,b2}, 36+5g ggnn{W,Wih,Whh,bih,bhh},
  // 51..55 att{Wa,ba,v,Wo,bo}, 56..59 enc{Wmu,bmu,Wvar,bvar}
  const float* xin[5] = {(const float*)d_in[0],(const float*)d_in[1],(const float*)d_in[2],
                         (const float*)d_in[3],(const float*)d_in[4]};
  const int cins[5] = {12,14,11,12,12};
  const int* nis[5] = {(const int*)d_in[6],(const int*)d_in[7],(const int*)d_in[8],
                       (const int*)d_in[9],(const int*)d_in[10]};
  const int* eis[3] = {(const int*)d_in[12],(const int*)d_in[13],(const int*)d_in[14]};
  const float* embed = (const float*)d_in[15];
  const float* Wa   = (const float*)d_in[51];
  const float* ba   = (const float*)d_in[52];
  const float* vatt = (const float*)d_in[53];
  const float* Wo   = (const float*)d_in[54];
  const float* bo   = (const float*)d_in[55];
  const float* Wmu  = (const float*)d_in[56];
  const float* bmu  = (const float*)d_in[57];
  const float* Wvar = (const float*)d_in[58];
  const float* bvar = (const float*)d_in[59];
  float* out = (float*)d_out;

  float* ws = (float*)d_ws;
  size_t off = 0;
  float* dfeat = ws + off; off += (size_t)NN*96;
  float* hb[3]; for (int i = 0; i < 3; ++i){ hb[i] = ws + off; off += (size_t)NN*GG; }
  float* mbuf = ws + off; off += (size_t)NN*GG;
  float* aggb = ws + off; off += (size_t)NN*GG;
  float* gib  = ws + off; off += (size_t)NN*384;
  float* ghb  = ws + off; off += (size_t)NN*384;
  unsigned* keys = (unsigned*)(ws + off); off += EE;
  int* ksrc = (int*)(ws + off); off += KEEPN;
  int* kdst = (int*)(ws + off); off += KEEPN;
  float* cnt = ws + off; off += NN;
  float* wt    = ws + off; off += (size_t)6*128*128;  // ggnn W^T, slot (gp-1)*3+layer
  float* waT   = ws + off; off += 128*128;
  float* woT   = ws + off; off += 128*128;
  float* wmuT  = ws + off; off += 64*128;
  float* wvarT = ws + off; off += 64*128;
  unsigned* st = (unsigned*)(ws + off); off += 512;

  auto grid1 = [](long n){ return (int)((n + 255)/256); };

  // --- weight transposes (Wih/Whh are already [out,K] = B^T, no transpose) ---
  for (int g = 1; g <= 2; ++g){
    const float* Wg = (const float*)d_in[36 + 5*g];
    for (int l = 0; l < 3; ++l)
      k_transpose<<<grid1(128*128), 256, 0, stream>>>(
          Wg + (size_t)l*128*128, wt + (size_t)((g-1)*3 + l)*128*128, 128, 128);
  }
  k_transpose<<<grid1(128*128), 256, 0, stream>>>(Wa,   waT,   128, 128);
  k_transpose<<<grid1(128*128), 256, 0, stream>>>(Wo,   woT,   128, 128);
  k_transpose<<<grid1(128*64),  256, 0, stream>>>(Wmu,  wmuT,  128, 64);
  k_transpose<<<grid1(128*64),  256, 0, stream>>>(Wvar, wvarT, 128, 64);

  // --- TCN branches scatter into d[N,96] ---
  for (int i = 0; i < 5; ++i){
    const float* w1 = (const float*)d_in[16 + 4*i + 0];
    const float* b1 = (const float*)d_in[16 + 4*i + 1];
    const float* w2 = (const float*)d_in[16 + 4*i + 2];
    const float* b2 = (const float*)d_in[16 + 4*i + 3];
    k_tcn<<<GN, 32, 0, stream>>>(xin[i], cins[i], w1, b1, w2, b2, nis[i], embed, dfeat);
  }

  auto gemm = [&](const float* A, const float* Bt, const float* bias, float* C,
                  int Ncols, int act){
    int blocks = (NN/128) * (Ncols/64);      // M=10880 = 85*128 exactly
    k_gemm_wmma<<<blocks, 256, 0, stream>>>(A, Bt, bias, C, Ncols, act);
  };

  // --- 3 GGNN runs: internal->ggnn[1], resource->ggnn[2], latent->ggnn[2] ---
  const int gset[3] = {1, 2, 2};
  for (int s = 0; s < 3; ++s){
    int gp = gset[s];
    const float* Wih = (const float*)d_in[37 + 5*gp];
    const float* Whh = (const float*)d_in[38 + 5*gp];
    const float* bih = (const float*)d_in[39 + 5*gp];
    const float* bhh = (const float*)d_in[40 + 5*gp];

    // structure learning: scores -> radix-select threshold -> compact kept set
    k_score<<<grid1(EE), 256, 0, stream>>>(dfeat, eis[s], keys);
    k_state_init<<<1, 512, 0, stream>>>(st);
    for (int p = 0; p < 4; ++p){
      k_hist<<<grid1(EE), 256, 0, stream>>>(keys, st, p);
      k_pick<<<1, 1, 0, stream>>>(st, p);
    }
    k_count_gt<<<grid1(EE), 256, 0, stream>>>(keys, st);
    k_finalize<<<1, 1, 0, stream>>>(st);
    k_fill_f32<<<grid1(NN), 256, 0, stream>>>(cnt, NN, 0.f);
    k_select<<<grid1(EE), 256, 0, stream>>>(keys, eis[s], st, ksrc, kdst, cnt);

    // GGNN layers
    float* h = hb[s];
    k_init_h<<<grid1((long)NN*GG), 256, 0, stream>>>(dfeat, h);
    for (int l = 0; l < 3; ++l){
      gemm(h, wt + (size_t)((gp-1)*3 + l)*128*128, nullptr, mbuf, 128, 0);   // m = h@W
      k_fill_f32<<<grid1((long)NN*GG), 256, 0, stream>>>(aggb, NN*GG, 0.f);
      k_aggregate<<<grid1((long)KEEPN*32), 256, 0, stream>>>(mbuf, ksrc, kdst, aggb);
      k_divide<<<grid1((long)NN*GG), 256, 0, stream>>>(aggb, cnt);           // mean
      gemm(aggb, Wih, bih, gib, 384, 0);                                     // gi
      gemm(h,    Whh, bhh, ghb, 384, 0);                                     // gh
      k_gate<<<grid1((long)NN*GG), 256, 0, stream>>>(gib, ghb, h);           // GRU gate
    }
  }

  // --- attention + output heads ---
  float* tb[3] = {gib, gib + (size_t)NN*GG, gib + (size_t)2*NN*GG};  // reuse gi space
  for (int s = 0; s < 3; ++s) gemm(hb[s], waT, ba, tb[s], 128, 1);   // tanh(h@Wa+ba)
  k_att<<<(NN + 3)/4, 128, 0, stream>>>(tb[0], tb[1], tb[2], hb[0], hb[1], hb[2],
                                        vatt, mbuf);                 // ctx in mbuf
  gemm(mbuf, woT, bo, aggb, 128, 0);                                 // hh in aggb
  gemm(aggb, wmuT,  bmu,  out,                 64, 0);               // z_mu
  gemm(aggb, wvarT, bvar, out + (size_t)NN*ZD, 64, 2);               // softplus z_var
}
